// LiquidTimeConstantNetwork_63574105915629
// MI455X (gfx1250) — matrix operand order repaired and checked by reading, not compiled, not run on hardware
//
#include <hip/hip_runtime.h>
#include <hip/hip_bf16.h>

typedef __attribute__((ext_vector_type(16))) __bf16 v16bf;
typedef __attribute__((ext_vector_type(4)))  __bf16 v4bf;
typedef __attribute__((ext_vector_type(8)))  float  v8f;

#define VOCAB 30522
#define HIDDEN 32
#define BATCH 8
#define SEQ 1024
#define LTC_STEPS 5
#define STEP_SIZE 0.01f

#if __has_builtin(__builtin_amdgcn_tanhf)
#define FAST_TANH(x) __builtin_amdgcn_tanhf(x)
#else
#define FAST_TANH(x) tanhf(x)
#endif

// Broadcast lane j's value to all lanes via v_readlane_b32 (SGPR result; no
// LDS round trip like ds_bpermute). j must be a compile-time constant here.
__device__ __forceinline__ float lane_bcast(float v, int j) {
    return __builtin_bit_cast(float,
        __builtin_amdgcn_readlane(__builtin_bit_cast(int, v), j));
}

// ---------------------------------------------------------------------------
// Phase 0: convert W_out (f32) -> bf16 once; reused 512x by the GEMM.
// 4 elements/thread: b128 load, b64 store.
// ---------------------------------------------------------------------------
__global__ void cvt_wout_kernel(const float4* __restrict__ w,
                                v4bf* __restrict__ o, int n4) {
    int i = blockIdx.x * blockDim.x + threadIdx.x;
    if (i < n4) {
        float4 v = w[i];
        v4bf r;
        r[0] = (__bf16)v.x; r[1] = (__bf16)v.y;
        r[2] = (__bf16)v.z; r[3] = (__bf16)v.w;
        o[i] = r;
    }
}

// ---------------------------------------------------------------------------
// Phase 1: nonlinear LTC scan (inherently sequential; latency-bound).
// One wave per batch row; lane i owns hidden unit i with W_rec[i][:],
// W_ad[i][:] register-resident. h_j broadcast by v_readlane; 4-way split
// accumulators shorten the FMA dependency chain; next-step embedding load
// is issued before the substep loop to hide its latency.
// ---------------------------------------------------------------------------
__global__ void ltc_scan_kernel(const int*   __restrict__ tokens,
                                const float* __restrict__ emb,
                                const float* __restrict__ Wrec,
                                const float* __restrict__ brec,
                                const float* __restrict__ Wad,
                                const float* __restrict__ bad,
                                const float* __restrict__ tau,
                                __bf16* __restrict__ hs_bf16) {
    const int b = blockIdx.x;     // batch row
    const int i = threadIdx.x;    // lane == hidden index (blockDim.x == 32)

    float wrec[HIDDEN], wad[HIDDEN];
#pragma unroll
    for (int j = 0; j < HIDDEN; ++j) {
        wrec[j] = Wrec[i * HIDDEN + j];
        wad[j]  = Wad [i * HIDDEN + j];
    }
    const float br = brec[i];
    const float ba = bad[i];
    const float tv_i = tau[i];

    const int* tok = tokens + b * SEQ;
    __bf16* outp = hs_bf16 + (size_t)b * SEQ * HIDDEN + i;

    float h = 0.0f;
    float u = emb[(size_t)tok[0] * HIDDEN + i];
    for (int t = 0; t < SEQ; ++t) {
        // issue next embedding load now; resolves under ~5 substeps of ALU
        float u_next = 0.0f;
        if (t + 1 < SEQ)
            u_next = emb[(size_t)tok[t + 1] * HIDDEN + i];

#pragma unroll
        for (int s = 0; s < LTC_STEPS; ++s) {
            float s0 = u + br, s1 = 0.f, s2 = 0.f, s3 = 0.f;
            float a0 = ba,     a1 = 0.f, a2 = 0.f, a3 = 0.f;
#pragma unroll
            for (int j = 0; j < HIDDEN; j += 4) {
                float h0 = lane_bcast(h, j + 0);
                float h1 = lane_bcast(h, j + 1);
                float h2 = lane_bcast(h, j + 2);
                float h3 = lane_bcast(h, j + 3);
                s0 = fmaf(h0, wrec[j + 0], s0);
                s1 = fmaf(h1, wrec[j + 1], s1);
                s2 = fmaf(h2, wrec[j + 2], s2);
                s3 = fmaf(h3, wrec[j + 3], s3);
                a0 = fmaf(h0, wad[j + 0], a0);
                a1 = fmaf(h1, wad[j + 1], a1);
                a2 = fmaf(h2, wad[j + 2], a2);
                a3 = fmaf(h3, wad[j + 3], a3);
            }
            const float S = (s0 + s1) + (s2 + s3);
            const float A = (a0 + a1) + (a2 + a3);
            const float tv = tv_i * FAST_TANH(S + A);
            h = fmaf(STEP_SIZE * (S - h), __builtin_amdgcn_rcpf(tv), h);
        }
        outp[(size_t)t * HIDDEN] = (__bf16)h;
        u = u_next;
    }
}

// ---------------------------------------------------------------------------
// Phase 2: out[8192, 30522] = hs[8192,32] @ W_out^T + b_out.
// K = 32 -> exactly ONE v_wmma_f32_16x16x32_bf16 per 16x16 output tile.
// Output (1.0 GB) is write-once: non-temporal stores keep L2 for A/B reuse.
// ---------------------------------------------------------------------------
__global__ void out_gemm_kernel(const __bf16* __restrict__ hs,     // [8192][32]
                                const __bf16* __restrict__ woutb,  // [30522][32]
                                const float*  __restrict__ bout,   // [30522]
                                float* __restrict__ out) {         // [8192][30522]
    const int lane  = threadIdx.x & 31;
    const int wave  = threadIdx.x >> 5;
    const int ntile = blockIdx.x * (blockDim.x >> 5) + wave;
    const int n0    = ntile * 16;
    if (n0 >= VOCAB) return;                 // ragged edge: whole wave retires
    const int m0    = blockIdx.y * 16;

    const int half = lane >> 4;              // 0: lanes 0-15, 1: lanes 16-31
    const int lm   = lane & 15;

    union { v16bf v; uint4 q[2]; } A, B;

    // A operand (16-bit A 16x32 layout): lane row m = m0+lm.
    // half==0 -> K{0..7} @ +0,  K{16..23} @ +16
    // half==1 -> K{8..15} @ +8, K{24..31} @ +24
    const __bf16* arow = hs + (size_t)(m0 + lm) * HIDDEN;
    A.q[0] = *(const uint4*)(arow + half * 8);
    A.q[1] = *(const uint4*)(arow + 16 + half * 8);

    // B operand (32x16): B[k][n] = W_out[n][k]. Lane col n = n0+lm (clamped).
    // half==0 -> K{0..7} @ +0, K{16..23} @ +16; half==1 -> K{8..15} @ +8, K{24..31} @ +24.
    const int ncol = n0 + lm;
    const int nc   = ncol < VOCAB ? ncol : (VOCAB - 1);
    const __bf16* brow = woutb + (size_t)nc * HIDDEN;
    B.q[0] = *(const uint4*)(brow + half * 8);
    B.q[1] = *(const uint4*)(brow + 16 + half * 8);

    v8f c = {0.f, 0.f, 0.f, 0.f, 0.f, 0.f, 0.f, 0.f};
    v8f d = __builtin_amdgcn_wmma_f32_16x16x32_bf16(
        /*neg_a=*/false, A.v, /*neg_b=*/false, B.v,
        /*c_mod=*/(short)0, c, /*reuse_a=*/false, /*reuse_b=*/false);

    const float bias = bout[nc];
    if (ncol < VOCAB) {
        // D layout: VGPR r, lanes 0-15 -> row m0+r; lanes 16-31 -> row m0+r+8.
        float* orow = out + (size_t)(m0 + half * 8) * VOCAB + ncol;
#pragma unroll
        for (int r = 0; r < 8; ++r)
            __builtin_nontemporal_store(d[r] + bias, &orow[(size_t)r * VOCAB]);
    }
}

// ---------------------------------------------------------------------------
extern "C" void kernel_launch(void* const* d_in, const int* in_sizes, int n_in,
                              void* d_out, int out_size, void* d_ws, size_t ws_size,
                              hipStream_t stream) {
    (void)in_sizes; (void)n_in; (void)out_size; (void)ws_size;
    const int*   tokens = (const int*)  d_in[0];
    const float* emb    = (const float*)d_in[1];
    const float* Wrec   = (const float*)d_in[2];
    const float* brec   = (const float*)d_in[3];
    const float* Wad    = (const float*)d_in[4];
    const float* bad    = (const float*)d_in[5];
    const float* Wout   = (const float*)d_in[6];
    const float* bout   = (const float*)d_in[7];
    const float* tau    = (const float*)d_in[8];
    float* out = (float*)d_out;

    // Workspace layout: [0, 1.91MB) W_out bf16 ; [2MB, 2.5MB) hs bf16
    __bf16* woutb = (__bf16*)d_ws;
    __bf16* hsb   = (__bf16*)((char*)d_ws + (size_t)(2u << 20));

    const int n4 = (VOCAB * HIDDEN) / 4;    // 976704 / 4
    cvt_wout_kernel<<<(n4 + 255) / 256, 256, 0, stream>>>(
        (const float4*)Wout, (v4bf*)woutb, n4);

    ltc_scan_kernel<<<BATCH, 32, 0, stream>>>(tokens, emb, Wrec, brec,
                                              Wad, bad, tau, hsb);

    const int ntiles = (VOCAB + 15) / 16;               // 1908
    dim3 grid((ntiles + 7) / 8, (BATCH * SEQ) / 16);    // 239 x 512
    out_gemm_kernel<<<grid, 256, 0, stream>>>(hsb, woutb, bout, out);
}
